// SDEGenerator_89567247991039
// MI455X (gfx1250) — matrix operand order repaired and checked by reading, not compiled, not run on hardware
//
#include <hip/hip_runtime.h>
#include <hip/hip_bf16.h>
#include <hip/hip_fp16.h>
#include <math.h>

// ---------------------------------------------------------------------------
// Neural-SDE generator for MI455X (gfx1250, wave32, WMMA).
//
// Roofline: ~1.4e11 FLOP vs ~55 MB HBM traffic -> compute/latency bound on a
// 511-step sequential scan. Strategy: partition batch (1024) into 32 tiles of
// 32 rows; each workgroup keeps ALL MLP weights in LDS as f16 (~260 KB, fits
// only because CDNA5 has 320 KB LDS/WGP) and integrates the full trajectory
// with v_wmma_f32_16x16x32_f16 (dual accumulators to break WMMA RAW chains).
// The t-column of W1 is folded into a per-step bias so every GEMM has K in
// {64,128}. f*dt and einsum(g,dW) are fused into WMMA epilogues via
// ds_add_f32 atomics on the f32 state. Per-step 1KB noise slices are
// double-buffered into LDS with GLOBAL_LOAD_ASYNC_TO_LDS_B128 + ASYNCcnt
// (guarded; falls back to direct loads + global_prefetch). Readout fused.
// ---------------------------------------------------------------------------

typedef __attribute__((ext_vector_type(16))) _Float16 v16h;
typedef __attribute__((ext_vector_type(8)))  _Float16 v8h;
typedef __attribute__((ext_vector_type(8)))  float    v8f;
typedef int v4i __attribute__((__vector_size__(16)));   // matches builtin proto

#define B_TOT     1024
#define T_TOT     512
#define L_DIM     64
#define N_DIM     8
#define H_DIM     128
#define DI_DIM    32
#define DO_DIM    16
#define STEPS_OUT 64
#define MULT_OUT  8

#define BM        32          // batch rows per workgroup
#define NWAVES    8
#define NTHREADS  256

#if defined(__HIP_DEVICE_COMPILE__) && __has_builtin(__builtin_amdgcn_global_load_async_to_lds_b128)
#define HAVE_ASYNC_LDS 1
#else
#define HAVE_ASYNC_LDS 0
#endif

// ---------------- LDS layout (bytes) ----------------
constexpr unsigned OFF_WF1  = 0;                              // 64x128 f16 (col-major)
constexpr unsigned OFF_WF2  = OFF_WF1  + 64  * 128 * 2;       // 128x128
constexpr unsigned OFF_WF3  = OFF_WF2  + 128 * 128 * 2;       // 128x64
constexpr unsigned OFF_WG1  = OFF_WF3  + 128 * 64  * 2;       // 64x128
constexpr unsigned OFF_WG2  = OFF_WG1  + 64  * 128 * 2;       // 128x128
constexpr unsigned OFF_WG3  = OFF_WG2  + 128 * 128 * 2;       // 128x512
constexpr unsigned OFF_WR1  = OFF_WG3  + 128 * 512 * 2;       // 64x128
constexpr unsigned OFF_WR2  = OFF_WR1  + 64  * 128 * 2;       // 128x16
constexpr unsigned OFF_ZF32 = OFF_WR2  + 128 * 16  * 2;       // BM x 64 f32 state
constexpr unsigned OFF_ZF16 = OFF_ZF32 + BM * 64 * 4;         // BM x 64 f16 copy
constexpr unsigned OFF_H1   = OFF_ZF16 + BM * 64 * 2;         // BM x 128 f16
constexpr unsigned OFF_H2   = OFF_H1   + BM * 128 * 2;        // BM x 128 f16
constexpr unsigned OFF_DW   = OFF_H2   + BM * 128 * 2;        // BM x 8 f32 (scaled)
constexpr unsigned OFF_BIAS = OFF_DW   + BM * 8 * 4;          // 1744 f32
constexpr unsigned NBIAS    = 1744;
constexpr unsigned OFF_TS   = OFF_BIAS + NBIAS * 4;           // 512 f32
constexpr unsigned OFF_DWRAW= OFF_TS   + T_TOT * 4;           // 2 x 1KB raw dW bufs
constexpr unsigned SMEM_BYTES = OFF_DWRAW + 2 * BM * N_DIM * 4;
static_assert(SMEM_BYTES <= 320u * 1024u, "LDS over 320KB WGP budget");

// bias-block float offsets
constexpr int BO_BF1E  = 0;     // 128 per-step effective f bias
constexpr int BO_GB1E  = 128;   // 128 per-step effective g bias
constexpr int BO_FB2   = 256;   // 128
constexpr int BO_FB3   = 384;   // 64
constexpr int BO_GB2   = 448;   // 128
constexpr int BO_GB3   = 576;   // 512
constexpr int BO_RB1   = 1088;  // 128
constexpr int BO_RB2   = 1216;  // 16
constexpr int BO_FB1R  = 1232;  // 128 raw f_b1
constexpr int BO_GB1R  = 1360;  // 128 raw g_b1
constexpr int BO_FW1T  = 1488;  // 128 f_W1 row 0 (t row)
constexpr int BO_GW1T  = 1616;  // 128 g_W1 row 0

union U16 { v16h v; v8h h[2]; };

__device__ __forceinline__ float softplus_f(float x) {
  return (x > 20.f) ? x : log1pf(expf(x));
}

// ---- async copy of one contiguous 1KB dW slice into LDS (wave 0 only) ----
// 32 lanes x 16B x 2 instructions; tracked by ASYNCcnt (2 per slice).
__device__ __forceinline__ void async_copy_dw_slice(const float* gbase,
                                                    float* lbase, int lane) {
#if HAVE_ASYNC_LDS
  char* gp = (char*)gbase + lane * 16;   // C-style cast drops const
  char* lp = (char*)lbase + lane * 16;
  __builtin_amdgcn_global_load_async_to_lds_b128(
      (__attribute__((address_space(1))) v4i*)gp,
      (__attribute__((address_space(3))) v4i*)lp, 0, 0);
  __builtin_amdgcn_global_load_async_to_lds_b128(
      (__attribute__((address_space(1))) v4i*)(gp + 512),
      (__attribute__((address_space(3))) v4i*)(lp + 512), 0, 0);
#else
  (void)gbase; (void)lbase; (void)lane;
#endif
}

__device__ __forceinline__ void wait_async_le2() {
#if HAVE_ASYNC_LDS
#if __has_builtin(__builtin_amdgcn_s_wait_asynccnt)
  __builtin_amdgcn_s_wait_asynccnt(2);
#else
  asm volatile("s_wait_asynccnt 0x2" ::: "memory");
#endif
#endif
}

__device__ __forceinline__ void wait_async_le0() {
#if HAVE_ASYNC_LDS
#if __has_builtin(__builtin_amdgcn_s_wait_asynccnt)
  __builtin_amdgcn_s_wait_asynccnt(0);
#else
  asm volatile("s_wait_asynccnt 0x0" ::: "memory");
#endif
#endif
}

// A fragment: 16x32 f16, activations row-major (row stride = Kdim halves).
// ISA 7.12.2 layout: two contiguous 16B runs per lane -> ds_load_b128 x2.
__device__ __forceinline__ v16h load_a_frag(const _Float16* A, int Kdim,
                                            int m0, int k0, int lane) {
  const int m  = m0 + (lane & 15);
  const int hh = lane >> 4;
  const _Float16* p = A + m * Kdim + k0 + 8 * hh;
  U16 u;
  u.h[0] = *(const v8h*)(p);
  u.h[1] = *(const v8h*)(p + 16);
  return u.v;
}

// B fragment: 32x16 f16, weights column-major (column stride = Kdim halves).
__device__ __forceinline__ v16h load_b_frag(const _Float16* W, int Kdim,
                                            int k0, int n0, int lane) {
  const int n  = n0 + (lane & 15);
  const int kh = lane >> 4;
  const _Float16* p = W + n * Kdim + k0 + 8 * kh;
  U16 u;
  u.h[0] = *(const v8h*)(p);
  u.h[1] = *(const v8h*)(p + 16);
  return u.v;
}

// Dual-accumulator K-reduction: two independent WMMA chains (even/odd 32-wide
// k-slices) merged at the end -> hides WMMA RAW latency.
template <int KD>
__device__ __forceinline__ v8f wmma_kloop(const _Float16* A, const _Float16* W,
                                          int m0, int n0, int lane) {
  static_assert(KD % 64 == 0, "KD must be a multiple of 64");
  v8f acc0 = {}, acc1 = {};
#pragma unroll
  for (int k0 = 0; k0 < KD; k0 += 64) {
    v16h a0 = load_a_frag(A, KD, m0, k0, lane);
    v16h b0 = load_b_frag(W, KD, k0, n0, lane);
    acc0 = __builtin_amdgcn_wmma_f32_16x16x32_f16(false, a0, false, b0,
                                                  (short)0, acc0, false, false);
    v16h a1 = load_a_frag(A, KD, m0, k0 + 32, lane);
    v16h b1 = load_b_frag(W, KD, k0 + 32, n0, lane);
    acc1 = __builtin_amdgcn_wmma_f32_16x16x32_f16(false, a1, false, b1,
                                                  (short)0, acc1, false, false);
  }
  return acc0 + acc1;
}

// GEMM (BM x KD) @ (KD x NOUT) + bias, activation, store f16 row-major.
// ACT: 0 = softplus, 1 = relu.
template <int KD, int NOUT, int ACT>
__device__ __forceinline__ void gemm_bias_act_store(
    const _Float16* __restrict__ A, const _Float16* __restrict__ W,
    const float* __restrict__ bias, _Float16* __restrict__ Out,
    int wave, int lane) {
  constexpr int MT = BM / 16;
  constexpr int TILES = MT * (NOUT / 16);
  for (int tile = wave; tile < TILES; tile += NWAVES) {
    const int mt = tile & (MT - 1);
    const int nt = tile / MT;
    v8f acc = wmma_kloop<KD>(A, W, mt * 16, nt * 16, lane);
    const int ncol = nt * 16 + (lane & 15);
    const int mh   = lane >> 4;
    const float bv = bias[ncol];
#pragma unroll
    for (int v = 0; v < 8; ++v) {
      const int row = mt * 16 + 8 * mh + v;
      float x = acc[v] + bv;
      float y = (ACT == 0) ? softplus_f(x) : fmaxf(x, 0.f);
      Out[row * NOUT + ncol] = (_Float16)y;
    }
  }
}

// Drift layer 3: z += tanh(h2 @ Wf3 + b3) * dt  (fused, ds_add_f32)
__device__ __forceinline__ void gemm_f3_accum(
    const _Float16* __restrict__ A, const _Float16* __restrict__ W,
    const float* __restrict__ b3, float* __restrict__ zacc, float dt,
    int wave, int lane) {
  constexpr int TILES = (BM / 16) * (L_DIM / 16);  // 8
  for (int tile = wave; tile < TILES; tile += NWAVES) {
    const int mt = tile & 1, nt = tile >> 1;
    v8f acc = wmma_kloop<128>(A, W, mt * 16, nt * 16, lane);
    const int ncol = nt * 16 + (lane & 15);
    const int mh   = lane >> 4;
    const float bv = b3[ncol];
#pragma unroll
    for (int v = 0; v < 8; ++v) {
      const int row = mt * 16 + 8 * mh + v;
      atomicAdd(&zacc[row * L_DIM + ncol], tanhf(acc[v] + bv) * dt);
    }
  }
}

// Diffusion layer 3: z[:,l] += tanh(h2 @ Wg3 + b3)[:, l*8+n] * dW[:,n]
__device__ __forceinline__ void gemm_g3_accum(
    const _Float16* __restrict__ A, const _Float16* __restrict__ W,
    const float* __restrict__ b3, const float* __restrict__ dWs,
    float* __restrict__ zacc, int wave, int lane) {
  constexpr int NOUT = L_DIM * N_DIM;              // 512
  constexpr int TILES = (BM / 16) * (NOUT / 16);   // 64
  for (int tile = wave; tile < TILES; tile += NWAVES) {
    const int mt = tile & 1, nt = tile >> 1;
    v8f acc = wmma_kloop<128>(A, W, mt * 16, nt * 16, lane);
    const int ncol = nt * 16 + (lane & 15);
    const int l  = ncol >> 3;
    const int nn = ncol & 7;
    const int mh = lane >> 4;
    const float bv = b3[ncol];
#pragma unroll
    for (int v = 0; v < 8; ++v) {
      const int row = mt * 16 + 8 * mh + v;
      atomicAdd(&zacc[row * L_DIM + l],
                tanhf(acc[v] + bv) * dWs[row * N_DIM + nn]);
    }
  }
}

// Readout layer 2: (BM x 128) @ (128 x 16) + rb2 -> global (full + matched).
__device__ __forceinline__ void readout_l2_write(
    const _Float16* __restrict__ A, const _Float16* __restrict__ W,
    const float* __restrict__ rb2, float* __restrict__ out_full,
    float* __restrict__ out_matched, int b0, int tIdx, int wave, int lane) {
  for (int tile = wave; tile < (BM / 16); tile += NWAVES) {
    const int mt = tile;
    v8f acc = wmma_kloop<128>(A, W, mt * 16, 0, lane);
    const int ncol = lane & 15;
    const int mh   = lane >> 4;
    const float bv = rb2[ncol];
#pragma unroll
    for (int v = 0; v < 8; ++v) {
      const int row  = mt * 16 + 8 * mh + v;
      const int grow = b0 + row;
      const float val = acc[v] + bv;
      out_full[(size_t)grow * (T_TOT * DO_DIM) + (size_t)tIdx * DO_DIM + ncol] = val;
      if ((tIdx & (MULT_OUT - 1)) == 0)
        out_matched[(size_t)grow * (STEPS_OUT * DO_DIM) +
                    (size_t)(tIdx >> 3) * DO_DIM + ncol] = val;
    }
  }
}

// Stage a (K x N) row-major f32 global weight into column-major f16 LDS.
__device__ __forceinline__ void stage_w(const float* __restrict__ G,
                                        _Float16* __restrict__ L,
                                        int K, int N, int rowOff, int tid) {
  for (int i = tid; i < K * N; i += NTHREADS) {
    const int n = i / K;
    const int k = i - n * K;
    L[i] = (_Float16)G[(k + rowOff) * N + n];
  }
}

__global__ __launch_bounds__(NTHREADS, 1)
void sde_gen_kernel(
    const float* __restrict__ init_noise, const float* __restrict__ dw,
    const float* __restrict__ ts,         const float* __restrict__ emb_W,
    const float* __restrict__ emb_b,
    const float* __restrict__ f_W1, const float* __restrict__ f_b1,
    const float* __restrict__ f_W2, const float* __restrict__ f_b2,
    const float* __restrict__ f_W3, const float* __restrict__ f_b3,
    const float* __restrict__ g_W1, const float* __restrict__ g_b1,
    const float* __restrict__ g_W2, const float* __restrict__ g_b2,
    const float* __restrict__ g_W3, const float* __restrict__ g_b3,
    const float* __restrict__ r_W1, const float* __restrict__ r_b1,
    const float* __restrict__ r_W2, const float* __restrict__ r_b2,
    float* __restrict__ out_full, float* __restrict__ out_matched) {
  extern __shared__ char smem[];
  _Float16* Wf1  = (_Float16*)(smem + OFF_WF1);
  _Float16* Wf2  = (_Float16*)(smem + OFF_WF2);
  _Float16* Wf3  = (_Float16*)(smem + OFF_WF3);
  _Float16* Wg1  = (_Float16*)(smem + OFF_WG1);
  _Float16* Wg2  = (_Float16*)(smem + OFF_WG2);
  _Float16* Wg3  = (_Float16*)(smem + OFF_WG3);
  _Float16* Wr1  = (_Float16*)(smem + OFF_WR1);
  _Float16* Wr2  = (_Float16*)(smem + OFF_WR2);
  float*    zf32 = (float*)   (smem + OFF_ZF32);
  _Float16* zf16 = (_Float16*)(smem + OFF_ZF16);
  _Float16* h1   = (_Float16*)(smem + OFF_H1);
  _Float16* h2   = (_Float16*)(smem + OFF_H2);
  float*    dWs  = (float*)   (smem + OFF_DW);
  float*    bias = (float*)   (smem + OFF_BIAS);
  float*    tsb  = (float*)   (smem + OFF_TS);
  float*    dwraw0 = (float*) (smem + OFF_DWRAW);
  float*    dwraw1 = dwraw0 + BM * N_DIM;

  const int tid  = threadIdx.x;
  const int wave = tid >> 5;
  const int lane = tid & 31;
  const int b0   = blockIdx.x * BM;
  const size_t dwStride = (size_t)B_TOT * N_DIM;   // floats per step

  // ---- one-time staging: weights f16 col-major, biases f32, ts ----
  stage_w(f_W1, Wf1, 64, 128, 1, tid);   // skip t-row
  stage_w(f_W2, Wf2, 128, 128, 0, tid);
  stage_w(f_W3, Wf3, 128, 64, 0, tid);
  stage_w(g_W1, Wg1, 64, 128, 1, tid);
  stage_w(g_W2, Wg2, 128, 128, 0, tid);
  stage_w(g_W3, Wg3, 128, 512, 0, tid);
  stage_w(r_W1, Wr1, 64, 128, 0, tid);
  stage_w(r_W2, Wr2, 128, 16, 0, tid);
  for (int i = tid; i < 128; i += NTHREADS) {
    bias[BO_FB2  + i] = f_b2[i];
    bias[BO_GB2  + i] = g_b2[i];
    bias[BO_RB1  + i] = r_b1[i];
    bias[BO_FB1R + i] = f_b1[i];
    bias[BO_GB1R + i] = g_b1[i];
    bias[BO_FW1T + i] = f_W1[i];   // row 0 of (65,128)
    bias[BO_GW1T + i] = g_W1[i];
  }
  for (int i = tid; i < 64;  i += NTHREADS) bias[BO_FB3 + i] = f_b3[i];
  for (int i = tid; i < 16;  i += NTHREADS) bias[BO_RB2 + i] = r_b2[i];
  for (int i = tid; i < 512; i += NTHREADS) {
    bias[BO_GB3 + i] = g_b3[i];
    tsb[i] = ts[i];
  }
  // z0 = init_noise @ emb_W + emb_b  (tiny: 32x32x64 per WG, scalar MACs)
  for (int i = tid; i < BM * L_DIM; i += NTHREADS) {
    const int m = i >> 6, l = i & 63;
    float acc = emb_b[l];
#pragma unroll
    for (int k = 0; k < DI_DIM; ++k)
      acc += init_noise[(size_t)(b0 + m) * DI_DIM + k] * emb_W[k * L_DIM + l];
    zf32[i] = acc;
    zf16[i] = (_Float16)acc;
  }
#if HAVE_ASYNC_LDS
  // prime double buffer with noise slices for steps 0 and 1
  if (wave == 0) {
    async_copy_dw_slice(dw + (size_t)b0 * N_DIM,            dwraw0, lane);
    async_copy_dw_slice(dw + dwStride + (size_t)b0 * N_DIM, dwraw1, lane);
  }
#endif
  __syncthreads();

  // ---- readout at t-index 0 ----
  gemm_bias_act_store<64, 128, 1>(zf16, Wr1, bias + BO_RB1, h1, wave, lane);
  __syncthreads();
  readout_l2_write(h1, Wr2, bias + BO_RB2, out_full, out_matched, b0, 0, wave, lane);
  __syncthreads();

  // ---- 511 Euler-Maruyama steps ----
  for (int s = 0; s < T_TOT - 1; ++s) {
    const float t  = tsb[s];
    const float dt = tsb[s + 1] - t;
    const float sq = sqrtf(dt);
    // per-step effective layer-1 biases: b1 + t * W1[0,:]
    if (tid < 128)
      bias[BO_BF1E + tid] = bias[BO_FB1R + tid] + t * bias[BO_FW1T + tid];
    else
      bias[BO_GB1E + tid - 128] = bias[BO_GB1R + tid - 128] + t * bias[BO_GW1T + tid - 128];
#if HAVE_ASYNC_LDS
    // ensure this step's raw noise slice has landed in LDS (wave-scoped cnt)
    if (wave == 0) {
      if (s == T_TOT - 2) wait_async_le0(); else wait_async_le2();
    }
#else
    {
      const int m = tid >> 3, nn = tid & 7;
      dWs[tid] = dw[(size_t)s * dwStride + (size_t)(b0 + m) * N_DIM + nn] * sq;
    }
    if (s + 1 < T_TOT - 1)
      __builtin_prefetch(&dw[(size_t)(s + 1) * dwStride +
                             (size_t)(b0 + (tid >> 3)) * N_DIM + (tid & 7)], 0, 0);
#endif
    __syncthreads();
#if HAVE_ASYNC_LDS
    dWs[tid] = ((s & 1) ? dwraw1 : dwraw0)[tid] * sq;   // scale raw slice
#endif

    // drift branch
    gemm_bias_act_store<64, 128, 0>(zf16, Wf1, bias + BO_BF1E, h1, wave, lane);
    __syncthreads();
#if HAVE_ASYNC_LDS
    // raw buffer for step s fully consumed (barrier above) -> refill for s+2
    if (wave == 0 && s + 2 < T_TOT - 1)
      async_copy_dw_slice(dw + (size_t)(s + 2) * dwStride + (size_t)b0 * N_DIM,
                          (s & 1) ? dwraw1 : dwraw0, lane);
#endif
    gemm_bias_act_store<128, 128, 0>(h1, Wf2, bias + BO_FB2, h2, wave, lane);
    __syncthreads();
    gemm_f3_accum(h2, Wf3, bias + BO_FB3, zf32, dt, wave, lane);
    __syncthreads();

    // diffusion branch
    gemm_bias_act_store<64, 128, 0>(zf16, Wg1, bias + BO_GB1E, h1, wave, lane);
    __syncthreads();
    gemm_bias_act_store<128, 128, 0>(h1, Wg2, bias + BO_GB2, h2, wave, lane);
    __syncthreads();
    gemm_g3_accum(h2, Wg3, bias + BO_GB3, dWs, zf32, wave, lane);
    __syncthreads();

    // refresh f16 state copy
    for (int i = tid; i < BM * L_DIM; i += NTHREADS) zf16[i] = (_Float16)zf32[i];
    __syncthreads();

    // fused readout at t-index s+1
    gemm_bias_act_store<64, 128, 1>(zf16, Wr1, bias + BO_RB1, h1, wave, lane);
    __syncthreads();
    readout_l2_write(h1, Wr2, bias + BO_RB2, out_full, out_matched, b0, s + 1, wave, lane);
    __syncthreads();
  }
}

extern "C" void kernel_launch(void* const* d_in, const int* in_sizes, int n_in,
                              void* d_out, int out_size, void* d_ws, size_t ws_size,
                              hipStream_t stream) {
  (void)in_sizes; (void)n_in; (void)out_size; (void)d_ws; (void)ws_size;
  const float* init_noise = (const float*)d_in[0];
  const float* dw_noise   = (const float*)d_in[1];
  const float* ts         = (const float*)d_in[2];
  const float* emb_W      = (const float*)d_in[3];
  const float* emb_b      = (const float*)d_in[4];
  const float* f_W1 = (const float*)d_in[5];
  const float* f_b1 = (const float*)d_in[6];
  const float* f_W2 = (const float*)d_in[7];
  const float* f_b2 = (const float*)d_in[8];
  const float* f_W3 = (const float*)d_in[9];
  const float* f_b3 = (const float*)d_in[10];
  const float* g_W1 = (const float*)d_in[11];
  const float* g_b1 = (const float*)d_in[12];
  const float* g_W2 = (const float*)d_in[13];
  const float* g_b2 = (const float*)d_in[14];
  const float* g_W3 = (const float*)d_in[15];
  const float* g_b3 = (const float*)d_in[16];
  const float* r_W1 = (const float*)d_in[17];
  const float* r_b1 = (const float*)d_in[18];
  const float* r_W2 = (const float*)d_in[19];
  const float* r_b2 = (const float*)d_in[20];

  float* out_full    = (float*)d_out;
  float* out_matched = out_full + (size_t)B_TOT * T_TOT * DO_DIM;

  // idempotent, deterministic host-side attribute set (no static guards)
  (void)hipFuncSetAttribute((const void*)sde_gen_kernel,
                            hipFuncAttributeMaxDynamicSharedMemorySize,
                            (int)SMEM_BYTES);

  sde_gen_kernel<<<dim3(B_TOT / BM), dim3(NTHREADS), SMEM_BYTES, stream>>>(
      init_noise, dw_noise, ts, emb_W, emb_b,
      f_W1, f_b1, f_W2, f_b2, f_W3, f_b3,
      g_W1, g_b1, g_W2, g_b2, g_W3, g_b3,
      r_W1, r_b1, r_W2, r_b2,
      out_full, out_matched);
}
